// SlotAttention_62895501083074
// MI455X (gfx1250) — matrix operand order repaired and checked by reading, not compiled, not run on hardware
//
#include <hip/hip_runtime.h>
#include <hip/hip_bf16.h>
#include <stdint.h>

#define BATCH 128
#define NTOK  4096
#define IND   64
#define DIMD  128
#define NSLOT 8
#define NITER 3
#define ATT_SCALE 0.08838834764831845f   /* 128^-0.5 */
#define EPSA  1e-8f
#define LNEPS 1e-5f

typedef __attribute__((ext_vector_type(16))) __bf16 bf16x16;
typedef __attribute__((ext_vector_type(8)))  __bf16 bf16x8;
typedef __attribute__((ext_vector_type(8)))  float  f32x8;

static __device__ __forceinline__ f32x8 wmma_bf16(bf16x16 a, bf16x16 b, f32x8 c) {
    // D = A(16x32 bf16) x B(32x16 bf16) + C(16x16 f32)
    return __builtin_amdgcn_wmma_f32_16x16x32_bf16(false, a, false, b, (short)0, c, false, false);
}
static __device__ __forceinline__ __bf16 f2b(float f) { return (__bf16)f; }

// ---------------------------------------------------------------------------
// Kernel 1: x = LN(inputs); k = x@Wk+bk (store bf16 row-major [b][j][d]);
//           v = x@Wv+bv (store bf16 transposed [b][d][j]).
// 8 waves/block, 16 tokens per wave, 128 tokens per block.
// ---------------------------------------------------------------------------
__global__ __launch_bounds__(256) void ln_kv_kernel(
    const float* __restrict__ inp, const float* __restrict__ Wk, const float* __restrict__ bk,
    const float* __restrict__ Wv, const float* __restrict__ bv,
    const float* __restrict__ g_in, const float* __restrict__ be_in,
    __bf16* __restrict__ kh, __bf16* __restrict__ vTh)
{
    __shared__ __align__(32) __bf16 wkT[DIMD * IND];   // [n][k] = Wk[k][n]
    __shared__ __align__(32) __bf16 wvT[DIMD * IND];
    __shared__ float sbk[DIMD], sbv[DIMD], sg[IND], sb[IND];

    const int tid = threadIdx.x;
    for (int idx = tid; idx < IND * DIMD; idx += 256) {
        int k = idx >> 7, n = idx & 127;
        wkT[n * IND + k] = f2b(Wk[idx]);
        wvT[n * IND + k] = f2b(Wv[idx]);
    }
    if (tid < DIMD) { sbk[tid] = bk[tid]; sbv[tid] = bv[tid]; }
    if (tid < IND)  { sg[tid] = g_in[tid]; sb[tid] = be_in[tid]; }
    __syncthreads();

    const int b     = blockIdx.x >> 5;
    const int jbase = (blockIdx.x & 31) * 128;
    const int wave  = tid >> 5;
    const int lane  = tid & 31;
    const int row   = lane & 15;     // token within tile / B column
    const int hi    = lane >> 4;
    const int j0    = jbase + wave * 16;

    // each lane holds 32 of the 64 input features of its row (A-layout groups)
    const size_t xbase = ((size_t)b * NTOK + j0 + row) * IND;
    float x[4][8];
    float s = 0.f, sq = 0.f;
#pragma unroll
    for (int g = 0; g < 4; ++g) {
        int koff = (g >> 1) * 32 + hi * 16 + (g & 1) * 8;
#pragma unroll
        for (int e = 0; e < 8; ++e) {
            float v = inp[xbase + koff + e];
            x[g][e] = v; s += v; sq += v * v;
        }
    }
    s  += __shfl_xor(s, 16);
    sq += __shfl_xor(sq, 16);
    float mean = s * (1.0f / 64.0f);
    float var  = sq * (1.0f / 64.0f) - mean * mean;
    float rstd = rsqrtf(var + LNEPS);

    bf16x16 a[2];   // A-layout: chunk c covers K = 32c..32c+31
#pragma unroll
    for (int g = 0; g < 4; ++g) {
        int koff = (g >> 1) * 32 + hi * 16 + (g & 1) * 8;
#pragma unroll
        for (int e = 0; e < 8; ++e) {
            float xn = (x[g][e] - mean) * rstd * sg[koff + e] + sb[koff + e];
            a[g >> 1][(g & 1) * 8 + e] = f2b(xn);
        }
    }

#pragma unroll
    for (int t = 0; t < 8; ++t) {
        const int n0 = t * 16;
        const int n  = n0 + row;              // output feature (B column)
        f32x8 acck, accv;
        float bkv = sbk[n], bvv = sbv[n];
#pragma unroll
        for (int r = 0; r < 8; ++r) { acck[r] = bkv; accv[r] = bvv; }
#pragma unroll
        for (int c = 0; c < 2; ++c) {
            bf16x16 bk16 = *(const bf16x16*)&wkT[n * IND + c * 32 + hi * 16];
            bf16x16 bv16 = *(const bf16x16*)&wvT[n * IND + c * 32 + hi * 16];
            acck = wmma_bf16(a[c], bk16, acck);
            accv = wmma_bf16(a[c], bv16, accv);
        }
        // k: [b][j][d]
#pragma unroll
        for (int r = 0; r < 8; ++r) {
            int m = r + 8 * hi;
            kh[((size_t)b * NTOK + j0 + m) * DIMD + n] = f2b(acck[r]);
        }
        // vT: [b][d][j], 8 consecutive tokens per lane -> one 16B store
        size_t vtb = ((size_t)b * DIMD + n) * NTOK + j0 + 8 * hi;
        bf16x8 vv;
#pragma unroll
        for (int r = 0; r < 8; ++r) vv[r] = f2b(accv[r]);
        *(bf16x8*)&vTh[vtb] = vv;
    }
}

// ---------------------------------------------------------------------------
// Kernel 2: s = LN(slots); q = s@Wq + bq -> bf16 [b][16][128] (rows 8..15 pad)
// ---------------------------------------------------------------------------
__global__ __launch_bounds__(256) void slot_q_kernel(
    const float* __restrict__ slots_cur, const float* __restrict__ Wq, const float* __restrict__ bq,
    const float* __restrict__ g_s, const float* __restrict__ be_s,
    __bf16* __restrict__ qh)
{
    __shared__ __align__(32) __bf16 sS[16 * DIMD];
    __shared__ __align__(32) __bf16 wqT[DIMD * DIMD];
    __shared__ float sbq[DIMD];
    const int tid = threadIdx.x;
    const int b   = blockIdx.x;

    for (int idx = tid; idx < 16 * DIMD; idx += 256) sS[idx] = f2b(0.0f);
    for (int idx = tid; idx < DIMD * DIMD; idx += 256) {
        int k = idx >> 7, n = idx & 127;
        wqT[n * DIMD + k] = f2b(Wq[idx]);
    }
    if (tid < DIMD) sbq[tid] = bq[tid];
    __syncthreads();

    const int wave = tid >> 5, lane = tid & 31;
    {   // LN of slot row `wave` (8 waves == 8 slots)
        const size_t base = ((size_t)b * NSLOT + wave) * DIMD;
        float xv[4]; float s = 0.f, sq = 0.f;
#pragma unroll
        for (int e = 0; e < 4; ++e) { float v = slots_cur[base + lane * 4 + e]; xv[e] = v; s += v; sq += v * v; }
#pragma unroll
        for (int sh = 1; sh < 32; sh <<= 1) { s += __shfl_xor(s, sh); sq += __shfl_xor(sq, sh); }
        float mean = s * (1.0f / 128.0f);
        float rstd = rsqrtf(sq * (1.0f / 128.0f) - mean * mean + LNEPS);
#pragma unroll
        for (int e = 0; e < 4; ++e) {
            int d = lane * 4 + e;
            sS[wave * DIMD + d] = f2b((xv[e] - mean) * rstd * g_s[d] + be_s[d]);
        }
    }
    __syncthreads();

    const int row = lane & 15, hi = lane >> 4;
    const int ncol = wave * 16 + row;
    f32x8 acc;
    float bqv = sbq[ncol];
#pragma unroll
    for (int r = 0; r < 8; ++r) acc[r] = bqv;
#pragma unroll
    for (int c = 0; c < 4; ++c) {
        bf16x16 av;
        bf16x8 lo = *(const bf16x8*)&sS[row * DIMD + c * 32 + hi * 16];
        bf16x8 h8 = *(const bf16x8*)&sS[row * DIMD + c * 32 + hi * 16 + 8];
#pragma unroll
        for (int e = 0; e < 8; ++e) { av[e] = lo[e]; av[8 + e] = h8[e]; }
        bf16x16 bv16 = *(const bf16x16*)&wqT[ncol * DIMD + c * 32 + hi * 16];
        acc = wmma_bf16(av, bv16, acc);
    }
#pragma unroll
    for (int r = 0; r < 8; ++r) {
        int m = r + 8 * hi;
        qh[((size_t)b * 16 + m) * DIMD + ncol] = f2b(acc[r]);
    }
}

// ---------------------------------------------------------------------------
// Kernel 3: dots = q@k^T * scale ; softmax over slots ; attn@v partials.
// Grid = B * 8 (each block covers 512 tokens; each wave owns 64 tokens).
// Deterministic: per-wave partials reduced in LDS in fixed order.
// ---------------------------------------------------------------------------
__global__ __launch_bounds__(256) void attn_kernel(
    const __bf16* __restrict__ qh, const __bf16* __restrict__ kh, const __bf16* __restrict__ vTh,
    float* __restrict__ U_part, float* __restrict__ rs_part)
{
    __shared__ __align__(32) __bf16 attn_lds[8][16 * 64]; // per wave: [slot][j-local]
    __shared__ float U_lds[8][16 * DIMD];                 // per-wave update partials
    __shared__ float rs_lds[8][8];

    const int b    = blockIdx.x >> 3;
    const int jblk = blockIdx.x & 7;
    const int tid  = threadIdx.x;
    const int wave = tid >> 5, lane = tid & 31;
    const int col  = lane & 15, hi = lane >> 4;
    const int jwb  = jblk * 512 + wave * 64;

    // q A-tiles (loop invariant): row = col (lane&15), K pattern per chunk
    bf16x16 qa[4];
    {
        const size_t qb = ((size_t)b * 16 + col) * DIMD;
#pragma unroll
        for (int c = 0; c < 4; ++c) {
            bf16x8 lo = *(const bf16x8*)&qh[qb + c * 32 + hi * 16];
            bf16x8 h8 = *(const bf16x8*)&qh[qb + c * 32 + hi * 16 + 8];
#pragma unroll
            for (int e = 0; e < 8; ++e) { qa[c][e] = lo[e]; qa[c][8 + e] = h8[e]; }
        }
    }

    float rs[8];
#pragma unroll
    for (int r = 0; r < 8; ++r) rs[r] = 0.f;

#pragma unroll
    for (int t = 0; t < 4; ++t) {
        const int j0 = jwb + t * 16;
        f32x8 dots;
#pragma unroll
        for (int r = 0; r < 8; ++r) dots[r] = 0.f;
#pragma unroll
        for (int c = 0; c < 4; ++c) {
            bf16x16 kb = *(const bf16x16*)&kh[((size_t)b * NTOK + j0 + col) * DIMD + c * 32 + hi * 16];
            dots = wmma_bf16(qa[c], kb, dots);
        }
        // softmax over the 8 real slots: C-layout puts rows 0..7 entirely in lanes 0-15
        float e8[8];
        float mx = -1e30f;
#pragma unroll
        for (int r = 0; r < 8; ++r) { e8[r] = dots[r] * ATT_SCALE; mx = fmaxf(mx, e8[r]); }
        float sum = 0.f;
#pragma unroll
        for (int r = 0; r < 8; ++r) { e8[r] = __expf(e8[r] - mx); sum += e8[r]; }
        float inv = 1.0f / sum;
#pragma unroll
        for (int r = 0; r < 8; ++r) {
            float av = (hi == 0) ? (e8[r] * inv + EPSA) : 0.0f;   // pad rows 8..15 -> 0
            rs[r] += av;
            attn_lds[wave][(r + 8 * hi) * 64 + t * 16 + col] = f2b(av);
        }
    }

    // rowsum partials: reduce across the 16 lanes holding real rows
#pragma unroll
    for (int r = 0; r < 8; ++r) {
        float v = rs[r];
        v += __shfl_xor(v, 1); v += __shfl_xor(v, 2);
        v += __shfl_xor(v, 4); v += __shfl_xor(v, 8);
        rs[r] = v;
    }
    if (lane == 0) {
#pragma unroll
        for (int r = 0; r < 8; ++r) rs_lds[wave][r] = rs[r];
    }

    // updates: U[i][d] += sum_j attn[i][j] * v[j][d]  (A = attn, B = vT rows)
    f32x8 U[8];
#pragma unroll
    for (int dt = 0; dt < 8; ++dt)
#pragma unroll
        for (int r = 0; r < 8; ++r) U[dt][r] = 0.f;

#pragma unroll
    for (int c2 = 0; c2 < 2; ++c2) {
        bf16x16 aa;
        bf16x8 lo = *(const bf16x8*)&attn_lds[wave][col * 64 + c2 * 32 + hi * 16];
        bf16x8 h8 = *(const bf16x8*)&attn_lds[wave][col * 64 + c2 * 32 + hi * 16 + 8];
#pragma unroll
        for (int e = 0; e < 8; ++e) { aa[e] = lo[e]; aa[8 + e] = h8[e]; }
#pragma unroll
        for (int dt = 0; dt < 8; ++dt) {
            const int d = dt * 16 + col;
            bf16x16 vb = *(const bf16x16*)&vTh[((size_t)b * DIMD + d) * NTOK + jwb + c2 * 32 + hi * 16];
            U[dt] = wmma_bf16(aa, vb, U[dt]);
        }
    }

#pragma unroll
    for (int dt = 0; dt < 8; ++dt)
#pragma unroll
        for (int r = 0; r < 8; ++r)
            U_lds[wave][(r + 8 * hi) * DIMD + dt * 16 + col] = U[dt][r];
    __syncthreads();

    float* Uout = U_part + (((size_t)b * 8 + jblk) * 16) * DIMD;
    for (int idx = tid; idx < 16 * DIMD; idx += 256) {
        float sm = 0.f;
#pragma unroll
        for (int w = 0; w < 8; ++w) sm += U_lds[w][idx];
        Uout[idx] = sm;
    }
    if (tid < 8) {
        float sm = 0.f;
#pragma unroll
        for (int w = 0; w < 8; ++w) sm += rs_lds[w][tid];
        rs_part[((size_t)b * 8 + jblk) * 8 + tid] = sm;
    }
}

// ---------------------------------------------------------------------------
// Kernel 4: updates = U/rowsum ; GRU ; LN ; residual MLP -> slots_cur
// (tiny: exact fp32 VALU)
// ---------------------------------------------------------------------------
__global__ __launch_bounds__(256) void gru_mlp_kernel(
    const float* __restrict__ U_part, const float* __restrict__ rs_part,
    const float* __restrict__ Wih, const float* __restrict__ Whh,
    const float* __restrict__ bih, const float* __restrict__ bhh,
    const float* __restrict__ W1, const float* __restrict__ b1,
    const float* __restrict__ W2, const float* __restrict__ b2,
    const float* __restrict__ g_p, const float* __restrict__ be_p,
    float* __restrict__ slots_cur)
{
    __shared__ float upd[NSLOT][DIMD];
    __shared__ float sprev[NSLOT][DIMD];
    __shared__ float gi[NSLOT][384];
    __shared__ float gh[NSLOT][384];
    __shared__ float snew[NSLOT][DIMD];
    __shared__ float pre[NSLOT][DIMD];
    __shared__ float hmlp[NSLOT][DIMD];

    const int b = blockIdx.x;
    const int tid = threadIdx.x;

    for (int idx = tid; idx < NSLOT * DIMD; idx += 256) {
        int i = idx >> 7, d = idx & 127;
        float sm = 0.f, rsum = 0.f;
#pragma unroll
        for (int jb = 0; jb < 8; ++jb) {
            sm   += U_part[(((size_t)b * 8 + jb) * 16 + i) * DIMD + d];
            rsum += rs_part[((size_t)b * 8 + jb) * 8 + i];
        }
        upd[i][d]   = sm / rsum;
        sprev[i][d] = slots_cur[((size_t)b * NSLOT + i) * DIMD + d];
    }
    __syncthreads();

    for (int idx = tid; idx < NSLOT * 384; idx += 256) {
        int i = idx / 384, o = idx % 384;
        const float* wi = Wih + (size_t)o * DIMD;
        const float* wh = Whh + (size_t)o * DIMD;
        float ai = 0.f, ah = 0.f;
        for (int d = 0; d < DIMD; ++d) { ai += upd[i][d] * wi[d]; ah += sprev[i][d] * wh[d]; }
        gi[i][o] = ai + bih[o];
        gh[i][o] = ah + bhh[o];
    }
    __syncthreads();

    for (int idx = tid; idx < NSLOT * DIMD; idx += 256) {
        int i = idx >> 7, d = idx & 127;
        float r  = 1.0f / (1.0f + __expf(-(gi[i][d]       + gh[i][d])));
        float z  = 1.0f / (1.0f + __expf(-(gi[i][128 + d] + gh[i][128 + d])));
        float nc = tanhf(gi[i][256 + d] + r * gh[i][256 + d]);
        snew[i][d] = (1.0f - z) * nc + z * sprev[i][d];
    }
    __syncthreads();

    {   // LN(snew) with g_p/be_p: wave w handles slot w
        const int wave = tid >> 5, lane = tid & 31;
        float xv[4]; float s = 0.f, sq = 0.f;
#pragma unroll
        for (int e = 0; e < 4; ++e) { float v = snew[wave][lane * 4 + e]; xv[e] = v; s += v; sq += v * v; }
#pragma unroll
        for (int sh = 1; sh < 32; sh <<= 1) { s += __shfl_xor(s, sh); sq += __shfl_xor(sq, sh); }
        float mean = s * (1.0f / 128.0f);
        float rstd = rsqrtf(sq * (1.0f / 128.0f) - mean * mean + LNEPS);
#pragma unroll
        for (int e = 0; e < 4; ++e) {
            int d = lane * 4 + e;
            pre[wave][d] = (xv[e] - mean) * rstd * g_p[d] + be_p[d];
        }
    }
    __syncthreads();

    for (int idx = tid; idx < NSLOT * DIMD; idx += 256) {
        int i = idx >> 7, o = idx & 127;
        float a = 0.f;
        for (int d = 0; d < DIMD; ++d) a += pre[i][d] * W1[(size_t)d * DIMD + o];
        hmlp[i][o] = fmaxf(a + b1[o], 0.f);
    }
    __syncthreads();

    for (int idx = tid; idx < NSLOT * DIMD; idx += 256) {
        int i = idx >> 7, d = idx & 127;
        float a = 0.f;
        for (int o = 0; o < DIMD; ++o) a += hmlp[i][o] * W2[(size_t)o * DIMD + d];
        slots_cur[((size_t)b * NSLOT + i) * DIMD + d] = snew[i][d] + a + b2[d];
    }
}

// ---------------------------------------------------------------------------
// Kernel 5: out = relu(slots@Wc1 + bc1) @ Wc2 + bc2   -> [B,8,64] f32
// ---------------------------------------------------------------------------
__global__ __launch_bounds__(256) void cast_out_kernel(
    const float* __restrict__ slots_cur,
    const float* __restrict__ Wc1, const float* __restrict__ bc1,
    const float* __restrict__ Wc2, const float* __restrict__ bc2,
    float* __restrict__ out)
{
    __shared__ float sl[NSLOT][DIMD];
    __shared__ float hc[NSLOT][64];
    const int b = blockIdx.x, tid = threadIdx.x;
    for (int idx = tid; idx < NSLOT * DIMD; idx += 256)
        sl[idx >> 7][idx & 127] = slots_cur[((size_t)b * NSLOT) * DIMD + idx];
    __syncthreads();
    for (int idx = tid; idx < NSLOT * 64; idx += 256) {
        int i = idx >> 6, c = idx & 63;
        float a = 0.f;
        for (int d = 0; d < DIMD; ++d) a += sl[i][d] * Wc1[(size_t)d * 64 + c];
        hc[i][c] = fmaxf(a + bc1[c], 0.f);
    }
    __syncthreads();
    for (int idx = tid; idx < NSLOT * 64; idx += 256) {
        int i = idx >> 6, o = idx & 63;
        float a = 0.f;
        for (int c = 0; c < 64; ++c) a += hc[i][c] * Wc2[(size_t)c * 64 + o];
        out[((size_t)b * NSLOT) * 64 + idx] = a + bc2[o];
    }
}

// ---------------------------------------------------------------------------
extern "C" void kernel_launch(void* const* d_in, const int* in_sizes, int n_in,
                              void* d_out, int out_size, void* d_ws, size_t ws_size,
                              hipStream_t stream) {
    const float* inputs = (const float*)d_in[0];
    const float* slots  = (const float*)d_in[1];
    const float* Wq  = (const float*)d_in[2];
    const float* bq  = (const float*)d_in[3];
    const float* Wk  = (const float*)d_in[4];
    const float* bk  = (const float*)d_in[5];
    const float* Wv  = (const float*)d_in[6];
    const float* bv  = (const float*)d_in[7];
    const float* Wih = (const float*)d_in[8];
    const float* Whh = (const float*)d_in[9];
    const float* bih = (const float*)d_in[10];
    const float* bhh = (const float*)d_in[11];
    const float* W1  = (const float*)d_in[12];
    const float* b1  = (const float*)d_in[13];
    const float* W2  = (const float*)d_in[14];
    const float* b2  = (const float*)d_in[15];
    const float* Wc1 = (const float*)d_in[16];
    const float* bc1 = (const float*)d_in[17];
    const float* Wc2 = (const float*)d_in[18];
    const float* bc2 = (const float*)d_in[19];
    const float* g_in = (const float*)d_in[20];
    const float* be_in = (const float*)d_in[21];
    const float* g_s  = (const float*)d_in[22];
    const float* be_s = (const float*)d_in[23];
    const float* g_p  = (const float*)d_in[24];
    const float* be_p = (const float*)d_in[25];

    // workspace layout (bytes)
    constexpr size_t KH_B  = (size_t)BATCH * NTOK * DIMD * 2;   // 134,217,728
    constexpr size_t VT_B  = KH_B;
    constexpr size_t QH_B  = (size_t)BATCH * 16 * DIMD * 2;    // 524,288
    constexpr size_t UP_B  = (size_t)BATCH * 8 * 16 * DIMD * 4; // 8,388,608
    constexpr size_t RS_B  = (size_t)BATCH * 8 * 8 * 4;         // 32,768
    char* ws = (char*)d_ws;
    __bf16* kh        = (__bf16*)(ws);
    __bf16* vTh       = (__bf16*)(ws + KH_B);
    __bf16* qh        = (__bf16*)(ws + KH_B + VT_B);
    float*  U_part    = (float*) (ws + KH_B + VT_B + QH_B);
    float*  rs_part   = (float*) (ws + KH_B + VT_B + QH_B + UP_B);
    float*  slots_cur = (float*) (ws + KH_B + VT_B + QH_B + UP_B + RS_B);

    hipMemcpyAsync(slots_cur, slots, (size_t)BATCH * NSLOT * DIMD * sizeof(float),
                   hipMemcpyDeviceToDevice, stream);

    ln_kv_kernel<<<BATCH * (NTOK / 128), 256, 0, stream>>>(
        inputs, Wk, bk, Wv, bv, g_in, be_in, kh, vTh);

    for (int it = 0; it < NITER; ++it) {
        slot_q_kernel<<<BATCH, 256, 0, stream>>>(slots_cur, Wq, bq, g_s, be_s, qh);
        attn_kernel<<<BATCH * 8, 256, 0, stream>>>(qh, kh, vTh, U_part, rs_part);
        gru_mlp_kernel<<<BATCH, 256, 0, stream>>>(U_part, rs_part, Wih, Whh, bih, bhh,
                                                  W1, b1, W2, b2, g_p, be_p, slots_cur);
    }
    cast_out_kernel<<<BATCH, 256, 0, stream>>>(slots_cur, Wc1, bc1, Wc2, bc2, (float*)d_out);
}
